// _SparseAttn_71640054497912
// MI455X (gfx1250) — compile-verified, matrix operand order checked by reading, not run on hardware
//
#include <hip/hip_runtime.h>
#include <math.h>

typedef __attribute__((ext_vector_type(16))) _Float16 v16h;
typedef __attribute__((ext_vector_type(8)))  _Float16 h8;
typedef __attribute__((ext_vector_type(2)))  _Float16 h2;
typedef __attribute__((ext_vector_type(8)))  float    v8f;

#define N_HEAD   16
#define HEAD_DIM 64
#define SEQ_T    2048
#define DMODEL   (N_HEAD * HEAD_DIM)   // 1024
#define BM       64                    // Q rows per workgroup
#define BN       64                    // K/V rows per iteration
#define NTILES   (SEQ_T / BN)          // 32
#define NWAVES   4
#define NTHREADS 128
#define NPAIRS   (HEAD_DIM / 2)        // 32 rope pairs per token

#define LOG2_THETA 13.287712379549449f
#define INV_2PI    0.15915494309189535f
#define TWO_PI     6.283185307179586f

// (token, pair) -> (cos, sin); 2048*32*8B = 512KB, L2-resident, shared by all
// heads/batches/q-tiles. Rebuilt every launch (deterministic, capture-safe).
__device__ float2 g_rope_tab[SEQ_T * NPAIRS];

__global__ void __launch_bounds__(256) rope_table_kernel() {
    int i   = blockIdx.x * 256 + threadIdx.x;  // [0, SEQ_T*NPAIRS)
    int tok = i >> 5;
    int d   = (i & 31) * 2;
    float freq = exp2f(-(float)d * (LOG2_THETA / (float)HEAD_DIM)) * INV_2PI;
    float r   = (float)tok * freq;
    float ang = (r - floorf(r)) * TWO_PI;
    float s, c;
    __sincosf(ang, &s, &c);
    float2 cs; cs.x = c; cs.y = s;
    g_rope_tab[i] = cs;
}

// ---- fragment gather from f16 LDS tile (row-major, stride 64) ---------------
// CDNA5 16-bit A/B fragment layout (ISA 7.12.2): per lane the 16 K-values are
// two contiguous 8-half runs -> exactly two ds_load_b128, no converts.
__device__ __forceinline__ v16h load_frag_h(const _Float16* lds, int row, int koff, int half) {
    const _Float16* base = lds + row * 64 + koff + 8 * half;
    h8 lo = *(const h8*)(base);
    h8 hi = *(const h8*)(base + 16);
    return __builtin_shufflevector(lo, hi, 0, 1, 2, 3, 4, 5, 6, 7,
                                           8, 9, 10, 11, 12, 13, 14, 15);
}

// ---- intra-16-lane XOR butterfly on the VALU pipe (v_permlane16_b32) --------
// selector nibble j (group-lane j) = j ^ m; same pattern for both 16-lane halves.
constexpr unsigned sel_lo(int m) {
    unsigned s = 0;
    for (int i = 0; i < 8; ++i) s |= (unsigned)((i ^ m) & 0xF) << (4 * i);
    return s;
}
constexpr unsigned sel_hi(int m) {
    unsigned s = 0;
    for (int i = 0; i < 8; ++i) s |= (unsigned)(((i + 8) ^ m) & 0xF) << (4 * i);
    return s;
}

#if __has_builtin(__builtin_amdgcn_permlane16)
#define XOR16(v, m)                                                             \
    __builtin_bit_cast(float, __builtin_amdgcn_permlane16(                      \
        __builtin_bit_cast(int, (v)), __builtin_bit_cast(int, (v)),             \
        (int)sel_lo(m), (int)sel_hi(m), false, false))
#else
#define XOR16(v, m) __shfl_xor((v), (m), 16)
#endif

__device__ __forceinline__ float red_max16(float v) {
    v = fmaxf(v, XOR16(v, 1));
    v = fmaxf(v, XOR16(v, 2));
    v = fmaxf(v, XOR16(v, 4));
    v = fmaxf(v, XOR16(v, 8));
    return v;
}
__device__ __forceinline__ float red_sum16(float v) {
    v += XOR16(v, 1);
    v += XOR16(v, 2);
    v += XOR16(v, 4);
    v += XOR16(v, 8);
    return v;
}

// ---- cooperative staging of one K/V tile into f16 LDS ----------------------
// K: rope'd via table, row-major [key][dim]; V: transposed [dim][key].
__device__ __forceinline__ void stage_kv(const float* __restrict__ Kb,
                                         const float* __restrict__ Vb,
                                         int kt, _Float16* skb, _Float16* svb,
                                         int tid) {
#pragma unroll
    for (int i = 0; i < 8; ++i) {
        int g   = tid + i * NTHREADS;  // 1024 groups of 4 elements
        int row = g >> 4;
        int d   = (g & 15) * 4;
        int tok = kt * BN + row;
        const float2* cs = &g_rope_tab[tok * NPAIRS + (d >> 1)];
        float2 cs0 = cs[0], cs1 = cs[1];
        float4 kv = *(const float4*)(Kb + (size_t)tok * DMODEL + d);
        h2 p0 = {(_Float16)(kv.x * cs0.x - kv.y * cs0.y),
                 (_Float16)(kv.y * cs0.x + kv.x * cs0.y)};
        h2 p1 = {(_Float16)(kv.z * cs1.x - kv.w * cs1.y),
                 (_Float16)(kv.w * cs1.x + kv.z * cs1.y)};
        h2* kd = (h2*)(skb + row * 64 + d);
        kd[0] = p0;
        kd[1] = p1;
        float4 vv = *(const float4*)(Vb + (size_t)tok * DMODEL + d);
        svb[(d + 0) * 64 + row] = (_Float16)vv.x;
        svb[(d + 1) * 64 + row] = (_Float16)vv.y;
        svb[(d + 2) * 64 + row] = (_Float16)vv.z;
        svb[(d + 3) * 64 + row] = (_Float16)vv.w;
    }
}

__global__ void __launch_bounds__(NTHREADS)
flash_rope_attn(const float* __restrict__ Q, const float* __restrict__ K,
                const float* __restrict__ V, float* __restrict__ O) {
    __shared__ _Float16 sQ[BM * 64];            // rope'd, pre-scaled Q tile (f16)
    __shared__ _Float16 sK[2][BN * 64];         // double-buffered K tiles (f16)
    __shared__ _Float16 sVt[2][64 * BN];        // double-buffered V^T tiles (f16)
    __shared__ _Float16 sP[NWAVES][16 * 64];    // per-wave P strip (f16)

    const int tid  = threadIdx.x;
    const int wave = tid >> 5;
    const int lane = tid & 31;
    const int half = lane >> 4;
    const int ln   = lane & 15;

    const int q0 = blockIdx.x * BM;
    const int h  = blockIdx.y;
    const int b  = blockIdx.z;

    const float* Qb = Q + (size_t)b * SEQ_T * DMODEL + h * HEAD_DIM;
    const float* Kb = K + (size_t)b * SEQ_T * DMODEL + h * HEAD_DIM;
    const float* Vb = V + (size_t)b * SEQ_T * DMODEL + h * HEAD_DIM;
    float*       Ob = O + (size_t)b * SEQ_T * DMODEL + h * HEAD_DIM;

    // ---- stage Q (rope via table + 1/sqrt(d) scale) and the first K/V tile -
#pragma unroll
    for (int i = 0; i < 8; ++i) {
        int g   = tid + i * NTHREADS;
        int row = g >> 4;
        int d   = (g & 15) * 4;
        int tok = q0 + row;
        const float2* cs = &g_rope_tab[tok * NPAIRS + (d >> 1)];
        float2 cs0 = cs[0], cs1 = cs[1];
        float4 qv = *(const float4*)(Qb + (size_t)tok * DMODEL + d);
        h2 p0 = {(_Float16)(0.125f * (qv.x * cs0.x - qv.y * cs0.y)),
                 (_Float16)(0.125f * (qv.y * cs0.x + qv.x * cs0.y))};
        h2 p1 = {(_Float16)(0.125f * (qv.z * cs1.x - qv.w * cs1.y)),
                 (_Float16)(0.125f * (qv.w * cs1.x + qv.z * cs1.y))};
        h2* qd = (h2*)(sQ + row * 64 + d);
        qd[0] = p0;
        qd[1] = p1;
    }
    stage_kv(Kb, Vb, 0, sK[0], sVt[0], tid);
    __syncthreads();

    // hoist per-wave Q A-fragments (reused for all 32 K tiles)
    const int mrow = 16 * wave;
    v16h qa0 = load_frag_h(sQ, mrow + ln, 0, half);
    v16h qa1 = load_frag_h(sQ, mrow + ln, 32, half);

    float mrun[8], lrun[8];
#pragma unroll
    for (int r = 0; r < 8; ++r) { mrun[r] = -1e30f; lrun[r] = 0.0f; }
    v8f acc[4];
#pragma unroll
    for (int t = 0; t < 4; ++t) acc[t] = v8f{};

    for (int kt = 0; kt < NTILES; ++kt) {
        const int cur = kt & 1;

        // overlap: stage next tile while computing on the current one
        if (kt + 1 < NTILES)
            stage_kv(Kb, Vb, kt + 1, sK[1 - cur], sVt[1 - cur], tid);
        if (kt + 2 < NTILES) {
            __builtin_prefetch(Kb + (size_t)((kt + 2) * BN + (tid >> 1)) * DMODEL, 0, 0);
            __builtin_prefetch(Vb + (size_t)((kt + 2) * BN + (tid >> 1)) * DMODEL, 0, 0);
        }

        const _Float16* kTile = sK[cur];
        const _Float16* vTile = sVt[cur];

        // ---- S = Q * K^T  (16 x 64 per wave) -------------------------------
        v8f st[4];
#pragma unroll
        for (int t = 0; t < 4; ++t) {
            v16h kb0 = load_frag_h(kTile, 16 * t + ln, 0, half);
            v16h kb1 = load_frag_h(kTile, 16 * t + ln, 32, half);
            v8f c = v8f{};
            c = __builtin_amdgcn_wmma_f32_16x16x32_f16(false, qa0, false, kb0, (short)0, c, false, false);
            c = __builtin_amdgcn_wmma_f32_16x16x32_f16(false, qa1, false, kb1, (short)0, c, false, false);
            st[t] = c;
        }

        // ---- online softmax ------------------------------------------------
#pragma unroll
        for (int r = 0; r < 8; ++r) {
            float vmax = fmaxf(fmaxf(st[0][r], st[1][r]), fmaxf(st[2][r], st[3][r]));
            vmax = red_max16(vmax);
            float mn    = fmaxf(mrun[r], vmax);
            float alpha = __expf(mrun[r] - mn);
            float rs = 0.0f;
#pragma unroll
            for (int t = 0; t < 4; ++t) {
                float p = __expf(st[t][r] - mn);
                st[t][r] = p;
                rs += p;
            }
            rs = red_sum16(rs);
            lrun[r] = lrun[r] * alpha + rs;
            mrun[r] = mn;
#pragma unroll
            for (int t = 0; t < 4; ++t) acc[t][r] *= alpha;
        }

        // ---- P: C-layout -> A-layout via per-wave LDS strip ----------------
        // Same-wave LDS ops complete in order (ISA 7.3) -> scheduling fence only.
        _Float16* pw = &sP[wave][0];
#pragma unroll
        for (int t = 0; t < 4; ++t)
#pragma unroll
            for (int r = 0; r < 8; ++r)
                pw[(r + 8 * half) * 64 + 16 * t + ln] = (_Float16)st[t][r];
        __builtin_amdgcn_wave_barrier();

        v16h pa0 = load_frag_h(&sP[wave][0], ln, 0, half);
        v16h pa1 = load_frag_h(&sP[wave][0], ln, 32, half);

        // ---- acc += P * V  -------------------------------------------------
#pragma unroll
        for (int t = 0; t < 4; ++t) {
            v16h vb0 = load_frag_h(vTile, 16 * t + ln, 0, half);
            v16h vb1 = load_frag_h(vTile, 16 * t + ln, 32, half);
            acc[t] = __builtin_amdgcn_wmma_f32_16x16x32_f16(false, pa0, false, vb0, (short)0, acc[t], false, false);
            acc[t] = __builtin_amdgcn_wmma_f32_16x16x32_f16(false, pa1, false, vb1, (short)0, acc[t], false, false);
        }

        __syncthreads();  // staging of next tile + all reads of current done
    }

    // ---- epilogue: normalize, write [B, T, H*N] ----------------------------
#pragma unroll
    for (int r = 0; r < 8; ++r) {
        float inv_l = 1.0f / lrun[r];
        int row = q0 + mrow + r + 8 * half;
#pragma unroll
        for (int t = 0; t < 4; ++t)
            Ob[(size_t)row * DMODEL + 16 * t + ln] = acc[t][r] * inv_l;
    }
}

extern "C" void kernel_launch(void* const* d_in, const int* in_sizes, int n_in,
                              void* d_out, int out_size, void* d_ws, size_t ws_size,
                              hipStream_t stream) {
    const float* Q = (const float*)d_in[0];
    const float* K = (const float*)d_in[1];
    const float* V = (const float*)d_in[2];
    float* O = (float*)d_out;
    (void)in_sizes; (void)n_in; (void)out_size; (void)d_ws; (void)ws_size;

    rope_table_kernel<<<(SEQ_T * NPAIRS) / 256, 256, 0, stream>>>();

    dim3 grid(SEQ_T / BM, N_HEAD, 2 /*B*/);
    dim3 block(NTHREADS);
    flash_rope_attn<<<grid, block, 0, stream>>>(Q, K, V, O);
}
